// mLSTMCell_46712064311922
// MI455X (gfx1250) — compile-verified, matrix operand order checked by reading, not run on hardware
//
#include <hip/hip_runtime.h>

// ---------------------------------------------------------------------------
// mLSTM cell for MI455X (gfx1250), bf16 WMMA (16x16x32) + TDM A-tile staging.
// D=512, HD=64, NH=8, B=4096.
//   out[0 .. 2097152)            : new_h (B x D) f32
//   out[2097152 .. 136314880)    : new_M (B x NH x HD x HD) f32
// ---------------------------------------------------------------------------

typedef __attribute__((ext_vector_type(16))) __bf16 v16bf;
typedef __attribute__((ext_vector_type(8)))  __bf16 v8bf;
typedef __attribute__((ext_vector_type(8)))  float  v8f;
typedef __attribute__((ext_vector_type(4)))  unsigned int v4u;
typedef __attribute__((ext_vector_type(8)))  int v8i;
typedef __attribute__((ext_vector_type(4)))  int v4i;

#define Dm   512
#define HD   64
#define NH   8
#define Bsz  4096

__device__ __forceinline__ __bf16 to_bf16(float f) {
    union { float f; unsigned u; } v; v.f = f;
    unsigned r = v.u + 0x7FFFu + ((v.u >> 16) & 1u);   // round-to-nearest-even
    unsigned short hs = (unsigned short)(r >> 16);
    __bf16 out;
    __builtin_memcpy(&out, &hs, 2);
    return out;
}

// One wave32 WMMA 16-bit fragment: lane holds K = {k0+8h+0..7, k0+16+8h+0..7}
// of one row (A) / one W-row (B).  `p` = row_base + k0 + 8*half (16B aligned).
__device__ __forceinline__ v16bf load_frag(const __bf16* p) {
    v8bf a = *(const v8bf*)(p);
    v8bf b = *(const v8bf*)(p + 16);
    v16bf f;
#pragma unroll
    for (int e = 0; e < 8; ++e) { f[e] = a[e]; f[8 + e] = b[e]; }
    return f;
}

// ---------------------------------------------------------------------------
__global__ void cvt_bf16_kernel(const float* __restrict__ src,
                                __bf16* __restrict__ dst, int n) {
    int i = blockIdx.x * blockDim.x + threadIdx.x;
    if (i < n) dst[i] = to_bf16(src[i]);
}

// ---------------------------------------------------------------------------
// C = act(A @ W.T + bias); software-pipelined bf16 WMMA.
// mode 0: exp(clip(s,-10,2))  mode 1: sigmoid  mode 2: identity
// block = 128 (4 waves), wave -> 16 rows x 64 cols, grid = (512/64, 4096/64)
// ---------------------------------------------------------------------------
__global__ void gate_gemm_kernel(const __bf16* __restrict__ A,
                                 const __bf16* __restrict__ W,
                                 const float*  __restrict__ bias,
                                 float* __restrict__ out, int mode) {
    const int lane  = threadIdx.x & 31;
    const int wave  = threadIdx.x >> 5;
    const int l16   = lane & 15;
    const int lhalf = lane >> 4;

    const int m0 = blockIdx.y * 64 + wave * 16;
    const int n0 = blockIdx.x * 64;

    const __bf16* arow = A + (size_t)(m0 + l16) * Dm + 8 * lhalf;
    const __bf16* wrow[4];
#pragma unroll
    for (int t = 0; t < 4; ++t)
        wrow[t] = W + (size_t)(n0 + t * 16 + l16) * Dm + 8 * lhalf;

    v8f acc[4] = {};

    // software pipeline: fragments for step k are loaded during step k-32
    v16bf a_cur = load_frag(arow);
    v16bf b_cur[4];
#pragma unroll
    for (int t = 0; t < 4; ++t) b_cur[t] = load_frag(wrow[t]);

    for (int k0 = 0; k0 < Dm; k0 += 32) {
        const int kn = (k0 + 32 < Dm) ? (k0 + 32) : k0;   // dummy reload on last
        v16bf a_nxt = load_frag(arow + kn);
        v16bf b_nxt[4];
#pragma unroll
        for (int t = 0; t < 4; ++t) b_nxt[t] = load_frag(wrow[t] + kn);
#pragma unroll
        for (int t = 0; t < 4; ++t)
            acc[t] = __builtin_amdgcn_wmma_f32_16x16x32_bf16(
                false, a_cur, false, b_cur[t], (short)0, acc[t], false, false);
        a_cur = a_nxt;
#pragma unroll
        for (int t = 0; t < 4; ++t) b_cur[t] = b_nxt[t];
    }

#pragma unroll
    for (int t = 0; t < 4; ++t) {
        const int col = n0 + t * 16 + l16;
        const float bcol = bias[col];
#pragma unroll
        for (int r = 0; r < 8; ++r) {
            const int row = m0 + r + 8 * lhalf;
            float s = acc[t][r] + bcol;
            float o;
            if (mode == 0)      o = __expf(fminf(fmaxf(s, -10.f), 2.f));
            else if (mode == 1) o = 1.f / (1.f + __expf(-s));
            else                o = s;
            out[(size_t)row * Dm + col] = o;
        }
    }
}

// ---------------------------------------------------------------------------
// Fused: v = x @ Wv.T + bv ; new_M = f^T*M + i^T*v ; Mout = new_M @ x_head ;
// gated = sigmoid_o * Mout (bf16).  One wave owns 16 batch rows x fixed g
// (g = head*64 + i) x all 64 j.  All 4 waves of a block share the same 16
// batch rows, so the 16x512 bf16 A tile (16KB) is staged in LDS once per
// block with the Tensor Data Mover, then read via ds_load.
// ---------------------------------------------------------------------------
__global__ void v_gemm_fused_kernel(const __bf16* __restrict__ xb,
                                    const __bf16* __restrict__ Wvb,
                                    const float*  __restrict__ bv,
                                    const float*  __restrict__ x,
                                    const float*  __restrict__ Min,
                                    const float*  __restrict__ ig,
                                    const float*  __restrict__ fg,
                                    const float*  __restrict__ og,
                                    float*  __restrict__ newM,
                                    __bf16* __restrict__ gated) {
    const int lane  = threadIdx.x & 31;
    const int wave  = threadIdx.x >> 5;
    const int l16   = lane & 15;
    const int lhalf = lane >> 4;

    const unsigned wid = blockIdx.x * 4u + wave;
    const int bt = wid >> 9;          // batch tile (16 rows); block-uniform
    const int g  = wid & 511;         // n*64 + i
    const int m0 = bt * 16;
    const int n  = g >> 6;            // head

    __shared__ __bf16 sA[16 * Dm];    // 16KB staged A tile (sole LDS alloc -> offset 0)

    // Prefetch the 16x64 f32 M tile consumed by the epilogue.
#pragma unroll
    for (int r = 0; r < 8; ++r) {
        const size_t brow = (size_t)(m0 + r + 8 * lhalf);
        __builtin_prefetch(&Min[brow * ((size_t)NH * HD * HD) + (size_t)g * 64 + l16 * 4], 0, 1);
    }

#if defined(__HIP_DEVICE_COMPILE__) && __has_builtin(__builtin_amdgcn_tensor_load_to_lds)
    if (wave == 0) {
        // Tensor DMA descriptor: 2D tile, 512 x 16 rows of 2B elements.
        const unsigned long long ga = (unsigned long long)(const void*)(xb + (size_t)m0 * Dm);
        v4u g0;
        g0[0] = 1u;                                     // count=1, user mode
        g0[1] = 0u;                                     // lds_addr = 0 (sA)
        g0[2] = (unsigned)(ga & 0xFFFFFFFFu);           // global_addr[31:0]
        g0[3] = (unsigned)((ga >> 32) & 0x1FFFFFFu)     // global_addr[56:32]
                | (2u << 30);                           // type=2 ("image")
        v8i g1;
        g1[0] = 0x10000;        // data_size=1 (2 bytes), no mask/iterate/pad
        g1[1] = (Dm << 16);     // tensor_dim0[15:0]=512
        g1[2] = (16 << 16);     // tensor_dim0[31:16]=0 | tensor_dim1[15:0]=16
        g1[3] = (Dm << 16);     // tensor_dim1[31:16]=0 | tile_dim0=512
        g1[4] = 16;             // tile_dim1=16, tile_dim2=0
        g1[5] = Dm;             // tensor_dim0_stride[31:0]=512
        g1[6] = 0;              // stride[47:32]=0, tensor_dim1_stride lo = 0
        g1[7] = 0;
        v4i z4 = {0, 0, 0, 0};  // groups 2/3 unused (2D tensor)
#if defined(__clang_major__) && __clang_major__ >= 23
        v8i z8 = {0, 0, 0, 0, 0, 0, 0, 0};
        __builtin_amdgcn_tensor_load_to_lds(g0, g1, z4, z4, z8, 0);
#else
        __builtin_amdgcn_tensor_load_to_lds(g0, g1, z4, z4, 0);
#endif
        __builtin_amdgcn_s_wait_tensorcnt(0);
    }
    __syncthreads();
#else
    // Fallback: cooperative 16B copies of the A tile.
    {
        const v8bf* src = (const v8bf*)(xb + (size_t)m0 * Dm);
        v8bf* dst = (v8bf*)sA;
        for (int idx = threadIdx.x; idx < 16 * Dm / 8; idx += blockDim.x)
            dst[idx] = src[idx];
        __syncthreads();
    }
#endif

    const __bf16* wrow[4];
#pragma unroll
    for (int t = 0; t < 4; ++t)
        wrow[t] = Wvb + (size_t)(g * 64 + t * 16 + l16) * Dm + 8 * lhalf;
    const __bf16* arow = sA + (size_t)l16 * Dm + 8 * lhalf;

    v8f acc[4] = {};

    v16bf b_cur[4];
#pragma unroll
    for (int t = 0; t < 4; ++t) b_cur[t] = load_frag(wrow[t]);

    for (int k0 = 0; k0 < Dm; k0 += 32) {
        const int kn = (k0 + 32 < Dm) ? (k0 + 32) : k0;
        v16bf b_nxt[4];
#pragma unroll
        for (int t = 0; t < 4; ++t) b_nxt[t] = load_frag(wrow[t] + kn);
        v16bf a = load_frag(arow + k0);              // ds_load from staged tile
#pragma unroll
        for (int t = 0; t < 4; ++t)
            acc[t] = __builtin_amdgcn_wmma_f32_16x16x32_bf16(
                false, a, false, b_cur[t], (short)0, acc[t], false, false);
#pragma unroll
        for (int t = 0; t < 4; ++t) b_cur[t] = b_nxt[t];
    }

    const size_t mstride = (size_t)NH * HD * HD;   // 32768 per batch row
#pragma unroll
    for (int r = 0; r < 8; ++r) {
        const int   brow = m0 + r + 8 * lhalf;
        const float fv = fg[(size_t)brow * Dm + g];
        const float iv = ig[(size_t)brow * Dm + g];
        float pr = 0.f;
#pragma unroll
        for (int t = 0; t < 4; ++t) {
            const int j = t * 16 + l16;
            const size_t midx = (size_t)brow * mstride + (size_t)g * 64 + j;
            const float vval = acc[t][r] + bv[g * 64 + j];
            const float nm = fv * Min[midx] + iv * vval;
            newM[midx] = nm;
            pr += nm * x[(size_t)brow * Dm + n * HD + j];
        }
        pr += __shfl_xor(pr, 1, 16);
        pr += __shfl_xor(pr, 2, 16);
        pr += __shfl_xor(pr, 4, 16);
        pr += __shfl_xor(pr, 8, 16);
        if (l16 == 0) {
            const float go = og[(size_t)brow * Dm + g];
            gated[(size_t)brow * Dm + g] = to_bf16(go * pr);
        }
    }
}

// ---------------------------------------------------------------------------
extern "C" void kernel_launch(void* const* d_in, const int* in_sizes, int n_in,
                              void* d_out, int out_size, void* d_ws, size_t ws_size,
                              hipStream_t stream) {
    const float* x    = (const float*)d_in[0];
    // d_in[1] = h (unused by the reference math)
    const float* M    = (const float*)d_in[2];
    const float* Wi   = (const float*)d_in[3];
    const float* bi   = (const float*)d_in[4];
    const float* Wf   = (const float*)d_in[5];
    const float* bfv  = (const float*)d_in[6];
    const float* Wo   = (const float*)d_in[7];
    const float* bo   = (const float*)d_in[8];
    const float* Wv   = (const float*)d_in[9];
    const float* bv   = (const float*)d_in[10];
    const float* Wout = (const float*)d_in[11];
    const float* bout = (const float*)d_in[12];

    char* ws = (char*)d_ws;
    __bf16* xb    = (__bf16*)(ws + 0);          //  4 MB  x bf16
    __bf16* Wvb   = (__bf16*)(ws + 4194304);    // 32 MB  Wv bf16 (L2-resident)
    __bf16* Wib   = (__bf16*)(ws + 37748736);
    __bf16* Wfb   = (__bf16*)(ws + 38273024);
    __bf16* Wob   = (__bf16*)(ws + 38797312);
    __bf16* Woutb = (__bf16*)(ws + 39321600);
    float*  igb   = (float*) (ws + 39845888);   //  8 MB  i_gate
    float*  fgb   = (float*) (ws + 48234496);   //  8 MB  f_gate
    float*  ogb   = (float*) (ws + 56623104);   //  8 MB  o_gate
    __bf16* gated = (__bf16*)(ws + 65011712);   //  4 MB  o*Mout bf16

    float* out_h = (float*)d_out;
    float* out_M = (float*)d_out + (size_t)Bsz * Dm;

    const int cb = 256;
    cvt_bf16_kernel<<<(Bsz * Dm + cb - 1) / cb, cb, 0, stream>>>(x, xb, Bsz * Dm);
    cvt_bf16_kernel<<<(Dm * HD * Dm + cb - 1) / cb, cb, 0, stream>>>(Wv, Wvb, Dm * HD * Dm);
    cvt_bf16_kernel<<<(Dm * Dm + cb - 1) / cb, cb, 0, stream>>>(Wi, Wib, Dm * Dm);
    cvt_bf16_kernel<<<(Dm * Dm + cb - 1) / cb, cb, 0, stream>>>(Wf, Wfb, Dm * Dm);
    cvt_bf16_kernel<<<(Dm * Dm + cb - 1) / cb, cb, 0, stream>>>(Wo, Wob, Dm * Dm);
    cvt_bf16_kernel<<<(Dm * Dm + cb - 1) / cb, cb, 0, stream>>>(Wout, Woutb, Dm * Dm);

    dim3 ggrid(Dm / 64, Bsz / 64);
    gate_gemm_kernel<<<ggrid, 128, 0, stream>>>(xb, Wib, bi,  igb, 0);
    gate_gemm_kernel<<<ggrid, 128, 0, stream>>>(xb, Wfb, bfv, fgb, 0);
    gate_gemm_kernel<<<ggrid, 128, 0, stream>>>(xb, Wob, bo,  ogb, 1);

    v_gemm_fused_kernel<<<(Bsz / 16) * 512 / 4, 128, 0, stream>>>(
        xb, Wvb, bv, x, M, igb, fgb, ogb, out_M, gated);

    gate_gemm_kernel<<<ggrid, 128, 0, stream>>>(gated, Woutb, bout, out_h, 2);
}
